// Head_11673721111224
// MI455X (gfx1250) — matrix operand order repaired and checked by reading, not compiled, not run on hardware
//
#include <hip/hip_runtime.h>
#include <hip/hip_bf16.h>

typedef __attribute__((ext_vector_type(16))) _Float16 v16h;
typedef __attribute__((ext_vector_type(8)))  _Float16 v8h;
typedef __attribute__((ext_vector_type(8)))  float    v8f;

#define BATCH  4
#define TSEQ   4096
#define CDIM   1024
#define HDIM   64
#define KSPLIT 4                     // waves per block, key-dim partitions

static __device__ __forceinline__ v8f vzero8() {
    v8f z;
#pragma unroll
    for (int i = 0; i < 8; ++i) z[i] = 0.0f;
    return z;
}

static __device__ __forceinline__ v8f wmma32(v16h a, v16h b, v8f c) {
    // D = A(16x32 f16) * B(32x16 f16) + C(16x16 f32)
    return __builtin_amdgcn_wmma_f32_16x16x32_f16(
        /*neg_a=*/false, a, /*neg_b=*/false, b,
        /*c_mod=*/(short)0, c, /*reuse_a=*/false, /*reuse_b=*/false);
}

// ---------------------------------------------------------------------------
// Prep: W [C,H] f32  ->  Wt [H,C] f16   (so WMMA B-frags are contiguous)
// ---------------------------------------------------------------------------
__global__ void wt_transpose_kernel(const float* __restrict__ W,
                                    _Float16* __restrict__ Wt) {
    int i = blockIdx.x * blockDim.x + threadIdx.x;
    if (i >= HDIM * CDIM) return;
    int h = i / CDIM;
    int c = i % CDIM;
    Wt[i] = (_Float16)W[c * HDIM + h];
}

// ---------------------------------------------------------------------------
// Fused QKV projection. One wave computes a 16x64 output tile of one of
// {q, k, v}. q is stored scaled by C^-0.5; v is stored transposed [B,H,T].
// ---------------------------------------------------------------------------
__global__ __launch_bounds__(32)
void qkv_proj_kernel(const float* __restrict__ x,
                     const _Float16* __restrict__ wtq,
                     const _Float16* __restrict__ wtk,
                     const _Float16* __restrict__ wtv,
                     _Float16* __restrict__ qb,
                     _Float16* __restrict__ kb,
                     _Float16* __restrict__ vtb) {
    const int mtile = blockIdx.x;          // 0 .. (B*T/16 - 1)
    const int mat   = blockIdx.y;          // 0=q 1=k 2=v
    const int lane  = threadIdx.x;
    const int col   = lane & 15;
    const int half  = lane >> 4;

    const _Float16* wt = (mat == 0) ? wtq : (mat == 1) ? wtk : wtv;

    v8f acc[4];
#pragma unroll
    for (int n = 0; n < 4; ++n) acc[n] = vzero8();

    const long rowbase = (long)mtile * 16;

    for (int kc = 0; kc < CDIM / 32; ++kc) {
        // A-frag from x (f32 -> f16): lane holds row col, K = half*16 .. +15
        // within this 32-wide K slice.
        const float* xp = x + (rowbase + col) * (long)CDIM + kc * 32 + half * 16;
        v8f c0 = *(const v8f*)(xp);        // K offs +0..7
        v8f c1 = *(const v8f*)(xp + 8);    // K offs +8..15
        v16h a;
#pragma unroll
        for (int j = 0; j < 8; ++j) {
            a[j]     = (_Float16)c0[j];
            a[j + 8] = (_Float16)c1[j];
        }
#pragma unroll
        for (int n = 0; n < 4; ++n) {
            // B-frag: lane holds column h = n*16+col, K = half*16 .. +15
            const v16h b = *(const v16h*)(wt + (n * 16 + col) * (long)CDIM +
                                          kc * 32 + half * 16);
            acc[n] = wmma32(a, b, acc[n]);
        }
    }

    // Store. C-layout: lane element r is row M = r + 8*half, col N = l%16.
    if (mat == 0) {
#pragma unroll
        for (int r = 0; r < 8; ++r) {
            long g = rowbase + r + 8 * half;
#pragma unroll
            for (int n = 0; n < 4; ++n)
                qb[g * HDIM + n * 16 + col] =
                    (_Float16)(acc[n][r] * 0.03125f);  // fold C^-0.5
        }
    } else if (mat == 1) {
#pragma unroll
        for (int r = 0; r < 8; ++r) {
            long g = rowbase + r + 8 * half;
#pragma unroll
            for (int n = 0; n < 4; ++n)
                kb[g * HDIM + n * 16 + col] = (_Float16)acc[n][r];
        }
    } else {
#pragma unroll
        for (int r = 0; r < 8; ++r) {
            long g = rowbase + r + 8 * half;
            long bb = g >> 12;         // / TSEQ
            long t  = g & (TSEQ - 1);
#pragma unroll
            for (int n = 0; n < 4; ++n)
                vtb[(bb * HDIM + n * 16 + col) * (long)TSEQ + t] =
                    (_Float16)acc[n][r];
        }
    }
}

// ---------------------------------------------------------------------------
// Flash attention with 4-way key split. A block of 4 waves owns 16 query
// rows; wave w streams keys [w*1024, (w+1)*1024) with private running
// (m, l, O), then the partials are flash-combined through LDS.
// ---------------------------------------------------------------------------
__global__ __launch_bounds__(32 * KSPLIT)
void flash_attn_kernel(const _Float16* __restrict__ qb,
                       const _Float16* __restrict__ kb,
                       const _Float16* __restrict__ vtb,
                       float* __restrict__ out) {
    const int tile = blockIdx.x;            // 0 .. 1023
    const int b    = tile >> 8;             // 256 query tiles per batch
    const int qi   = (tile & 255) * 16;
    const int tid  = threadIdx.x;
    const int wave = tid >> 5;
    const int lane = tid & 31;
    const int col  = lane & 15;
    const int half = lane >> 4;

    __shared__ float    Osh[KSPLIT][16][HDIM];
    __shared__ float    msh[KSPLIT][16];
    __shared__ float    lsh[KSPLIT][16];
    __shared__ _Float16 P[KSPLIT][16 * 32];
    _Float16* Pw = &P[wave][0];

    // q A-frags for the two 32-wide K(=H) slices; fixed for whole kernel.
    const _Float16* qrow = qb + ((long)(b * TSEQ + qi + col)) * HDIM;
    v16h aq[2];
#pragma unroll
    for (int hs = 0; hs < 2; ++hs) {
        v8h lo = *(const v8h*)(qrow + hs * 32 + half * 16);
        v8h hi = *(const v8h*)(qrow + hs * 32 + half * 16 + 8);
#pragma unroll
        for (int j = 0; j < 8; ++j) { aq[hs][j] = lo[j]; aq[hs][j + 8] = hi[j]; }
    }

    float m[8], lsum[8];
    v8f   O[4];
#pragma unroll
    for (int r = 0; r < 8; ++r) { m[r] = -1e30f; lsum[r] = 0.0f; }
#pragma unroll
    for (int n = 0; n < 4; ++n) O[n] = vzero8();

    const _Float16* kbase = kb  + (long)b * TSEQ * HDIM;
    const _Float16* vbase = vtb + (long)b * HDIM * TSEQ;

    const int keyBeg = wave * (TSEQ / KSPLIT);
    const int keyEnd = keyBeg + (TSEQ / KSPLIT);

    for (int key0 = keyBeg; key0 < keyEnd; key0 += 32) {
        // ---- scores: two 16x16 tiles over 32 keys, K-loop over H=64 ----
        v8f s0 = vzero8(), s1 = vzero8();
#pragma unroll
        for (int hs = 0; hs < 2; ++hs) {
            const v16h bk0 = *(const v16h*)(kbase + (long)(key0 + col) * HDIM +
                                            hs * 32 + half * 16);
            const v16h bk1 = *(const v16h*)(kbase + (long)(key0 + 16 + col) * HDIM +
                                            hs * 32 + half * 16);
            s0 = wmma32(aq[hs], bk0, s0);
            s1 = wmma32(aq[hs], bk1, s1);
        }

        // ---- online softmax update (rows live across 16-lane halves) ----
        float p0[8], p1[8];
#pragma unroll
        for (int r = 0; r < 8; ++r) {
            float t = fmaxf(s0[r], s1[r]);
#pragma unroll
            for (int mask = 1; mask < 16; mask <<= 1)
                t = fmaxf(t, __shfl_xor(t, mask, 32));
            float mn   = fmaxf(m[r], t);
            float corr = __expf(m[r] - mn);
            p0[r] = __expf(s0[r] - mn);
            p1[r] = __expf(s1[r] - mn);
            float rs = p0[r] + p1[r];
#pragma unroll
            for (int mask = 1; mask < 16; mask <<= 1)
                rs += __shfl_xor(rs, mask, 32);
            lsum[r] = lsum[r] * corr + rs;
            m[r]    = mn;
            O[0][r] *= corr; O[1][r] *= corr; O[2][r] *= corr; O[3][r] *= corr;
        }

        // ---- C-layout -> A-layout restripe of P through per-wave LDS ----
        // Same-wave LDS ops are in-order (DScnt), so no barrier is needed.
#pragma unroll
        for (int r = 0; r < 8; ++r) {
            Pw[(r + 8 * half) * 32 + col]      = (_Float16)p0[r];
            Pw[(r + 8 * half) * 32 + 16 + col] = (_Float16)p1[r];
        }
        v16h pa;
        {
            v8h lo = *(const v8h*)(&Pw[col * 32 + half * 16]);
            v8h hi = *(const v8h*)(&Pw[col * 32 + half * 16 + 8]);
#pragma unroll
            for (int j = 0; j < 8; ++j) { pa[j] = lo[j]; pa[j + 8] = hi[j]; }
        }

        // ---- O += P(16x32) @ V(32x64), B-frags contiguous from vT ----
#pragma unroll
        for (int n = 0; n < 4; ++n) {
            const v16h vb = *(const v16h*)(vbase + (long)(n * 16 + col) * TSEQ +
                                           key0 + half * 16);
            O[n] = wmma32(pa, vb, O[n]);
        }
    }

    // ---- publish per-wave partials ----
#pragma unroll
    for (int r = 0; r < 8; ++r) {
        int row = r + 8 * half;
        if (col == 0) { msh[wave][row] = m[r]; lsh[wave][row] = lsum[r]; }
#pragma unroll
        for (int n = 0; n < 4; ++n)
            Osh[wave][row][n * 16 + col] = O[n][r];
    }
    __syncthreads();

    // ---- flash combine across the 4 key partitions; 1024 outputs ----
#pragma unroll
    for (int i = 0; i < (16 * HDIM) / (32 * KSPLIT); ++i) {
        int idx = tid + i * (32 * KSPLIT);
        int row = idx >> 6;          // /HDIM
        int h   = idx & (HDIM - 1);
        float mstar = msh[0][row];
#pragma unroll
        for (int w = 1; w < KSPLIT; ++w) mstar = fmaxf(mstar, msh[w][row]);
        float num = 0.0f, den = 0.0f;
#pragma unroll
        for (int w = 0; w < KSPLIT; ++w) {
            float f = __expf(msh[w][row] - mstar);
            num += f * Osh[w][row][h];
            den += f * lsh[w][row];
        }
        out[((long)(b * TSEQ + qi + row)) * HDIM + h] = num / den;
    }
}

// ---------------------------------------------------------------------------
extern "C" void kernel_launch(void* const* d_in, const int* in_sizes, int n_in,
                              void* d_out, int out_size, void* d_ws, size_t ws_size,
                              hipStream_t stream) {
    const float* x  = (const float*)d_in[0];
    const float* Wq = (const float*)d_in[1];
    const float* Wk = (const float*)d_in[2];
    const float* Wv = (const float*)d_in[3];
    float* out = (float*)d_out;

    _Float16* wtq = (_Float16*)d_ws;                 // 64*1024
    _Float16* wtk = wtq + HDIM * CDIM;
    _Float16* wtv = wtk + HDIM * CDIM;
    _Float16* qb  = wtv + HDIM * CDIM;               // 16384*64 each
    _Float16* kb  = qb + (long)BATCH * TSEQ * HDIM;
    _Float16* vtb = kb + (long)BATCH * TSEQ * HDIM;

    const int nwt = HDIM * CDIM;
    wt_transpose_kernel<<<(nwt + 255) / 256, 256, 0, stream>>>(Wq, wtq);
    wt_transpose_kernel<<<(nwt + 255) / 256, 256, 0, stream>>>(Wk, wtk);
    wt_transpose_kernel<<<(nwt + 255) / 256, 256, 0, stream>>>(Wv, wtv);

    qkv_proj_kernel<<<dim3(BATCH * TSEQ / 16, 3), 32, 0, stream>>>(
        x, wtq, wtk, wtv, qb, kb, vtb);

    flash_attn_kernel<<<BATCH * TSEQ / 16, 32 * KSPLIT, 0, stream>>>(
        qb, kb, vtb, out);
}
